// DSAM_71717363908895
// MI455X (gfx1250) — matrix operand order repaired and checked by reading, not compiled, not run on hardware
//
#include <hip/hip_runtime.h>

// ---------------- types ----------------
typedef __attribute__((ext_vector_type(16))) __bf16 v16bf;
typedef __attribute__((ext_vector_type(8)))  __bf16 v8bf;
typedef __attribute__((ext_vector_type(8)))  float  v8f;
typedef __attribute__((ext_vector_type(4)))  float  v4f;

#define PIX   32768   // B*H*W
#define HWSZ  4096
#define BATCH 8
#define CIN   384
#define DCH   128
#define NRES  512
#define KTOP  81
#define LN_EPS 1e-5f

// ---------------- device helpers ----------------
__device__ __forceinline__ float wave_sum(float v) {
#pragma unroll
  for (int o = 16; o > 0; o >>= 1) v += __shfl_xor(v, o, 32);
  return v;
}
__device__ __forceinline__ float wave_max(float v) {
#pragma unroll
  for (int o = 16; o > 0; o >>= 1) v = fmaxf(v, __shfl_xor(v, o, 32));
  return v;
}

// A-matrix fragment (16x32 bf16, row-major source, leading dim ld).
// ISA layout: lane m (m=lane&15, h=lane>>4): elems 0..7 = K[8h..8h+7],
// elems 8..15 = K[16+8h..16+8h+7].
__device__ __forceinline__ v16bf load_frag_a(const __bf16* base, int ld, int lane) {
  int m = lane & 15, h = lane >> 4;
  const __bf16* p = base + (size_t)m * ld + h * 8;
  v8bf lo = *(const v8bf*)(p);
  v8bf hi = *(const v8bf*)(p + 16);
  return __builtin_shufflevector(lo, hi, 0,1,2,3,4,5,6,7,8,9,10,11,12,13,14,15);
}

// B-matrix fragment (32x16 bf16). Source holds column n of B as a contiguous
// row (row-major [n][k], leading dim ld). ISA layout: lane n (n=lane&15,
// h=lane>>4): elems 0..7 = K[8h..8h+7], elems 8..15 = K[16+8h..16+8h+7].
__device__ __forceinline__ v16bf load_frag_b(const __bf16* base, int ld, int lane) {
  int n = lane & 15, h = lane >> 4;
  const __bf16* p = base + (size_t)n * ld + h * 8;
  v8bf lo = *(const v8bf*)(p);
  v8bf hi = *(const v8bf*)(p + 16);
  return __builtin_shufflevector(lo, hi, 0,1,2,3,4,5,6,7,8,9,10,11,12,13,14,15);
}

__device__ __forceinline__ v8f wmma_bf16(v16bf a, v16bf b, v8f c) {
  return __builtin_amdgcn_wmma_f32_16x16x32_bf16(false, a, false, b, (short)0, c,
                                                 false, false);
}

// ---------------- prep kernels ----------------

// x [B,C,H,W] f32 -> xb [B*HW, C] bf16 (LDS tile transpose)
__global__ __launch_bounds__(256) void kx_transpose(const float* __restrict__ x,
                                                    __bf16* __restrict__ xb) {
  __shared__ float tile[32][33];
  int b = blockIdx.z, c0 = blockIdx.y * 32, hw0 = blockIdx.x * 32;
  int tx = threadIdx.x, ty = threadIdx.y;
#pragma unroll
  for (int i = 0; i < 4; i++) {
    int cl = ty + i * 8;
    tile[cl][tx] = x[((size_t)b * CIN + c0 + cl) * HWSZ + hw0 + tx];
  }
  __syncthreads();
#pragma unroll
  for (int i = 0; i < 4; i++) {
    int hl = ty + i * 8;
    xb[((size_t)b * HWSZ + hw0 + hl) * CIN + c0 + tx] = (__bf16)tile[tx][hl];
  }
}

// W2 = sem_w @ conv_in_w  (128x384), b2 = sem_w @ conv_in_b + sem_b
__global__ __launch_bounds__(256) void kw2(const float* __restrict__ semw,
                                           const float* __restrict__ cinw,
                                           const float* __restrict__ cinb,
                                           const float* __restrict__ semb,
                                           __bf16* __restrict__ w2,
                                           float* __restrict__ b2) {
  int id = blockIdx.x * 256 + threadIdx.x;
  if (id >= DCH * CIN) return;
  int o = id / CIN, c = id % CIN;
  float acc = 0.f;
  for (int m = 0; m < DCH; m++) acc += semw[o * DCH + m] * cinw[m * CIN + c];
  w2[id] = (__bf16)acc;
  if (c == 0) {
    float bb = semb[o];
    for (int m = 0; m < DCH; m++) bb += semw[o * DCH + m] * cinb[m];
    b2[o] = bb;
  }
}

// f32 -> bf16 convert (conv_out_w)
__global__ void kcvt(const float* __restrict__ w, __bf16* __restrict__ o, int n) {
  int id = blockIdx.x * 256 + threadIdx.x;
  if (id < n) o[id] = (__bf16)w[id];
}

// drn = LN(DR) rows (norm2 params); cdr[n] = ||drn[n]||; store drn and drnT bf16
__global__ __launch_bounds__(256) void kdrn(const float* __restrict__ DR,
                                            const float* __restrict__ w,
                                            const float* __restrict__ b,
                                            __bf16* __restrict__ drn,
                                            __bf16* __restrict__ drnT,
                                            float* __restrict__ cdr) {
  int lane = threadIdx.x & 31, wv = threadIdx.x >> 5;
  int n = blockIdx.x * 8 + wv;
  float v[4];
  float s = 0.f, sq = 0.f;
#pragma unroll
  for (int i = 0; i < 4; i++) {
    v[i] = DR[(size_t)n * DCH + lane + 32 * i];
    s += v[i]; sq += v[i] * v[i];
  }
  s = wave_sum(s); sq = wave_sum(sq);
  float mean = s * (1.f / DCH);
  float var = sq * (1.f / DCH) - mean * mean;
  float rstd = rsqrtf(var + LN_EPS);
  float ln[4]; float s2 = 0.f;
#pragma unroll
  for (int i = 0; i < 4; i++) {
    int c = lane + 32 * i;
    ln[i] = (v[i] - mean) * rstd * w[c] + b[c];
    s2 += ln[i] * ln[i];
  }
  s2 = wave_sum(s2);
  if (lane == 0) cdr[n] = sqrtf(s2);
#pragma unroll
  for (int i = 0; i < 4; i++) {
    int c = lane + 32 * i;
    __bf16 bv = (__bf16)ln[i];
    drn[(size_t)n * DCH + c] = bv;
    drnT[(size_t)c * NRES + n] = bv;
  }
}

// ---------------- K1: fused (conv_in∘sem) GEMM + double-LN ----------------
// 16 pixel-rows per workgroup; 8 waves each own one 16-col tile of D=128.
__global__ __launch_bounds__(256) void k1_gemm_ln(const __bf16* __restrict__ xb,
                                                  const __bf16* __restrict__ w2,
                                                  const float* __restrict__ b2,
                                                  const float* __restrict__ w1,
                                                  const float* __restrict__ b1,
                                                  __bf16* __restrict__ ln2out,
                                                  float* __restrict__ cdl) {
  __shared__ float xs[16][136];
  int lane = threadIdx.x & 31, wv = threadIdx.x >> 5;
  int row0 = blockIdx.x * 16;
  int n0 = wv * 16;
  v8f acc = {};
#pragma unroll
  for (int kk = 0; kk < 12; kk++) {
    v16bf a = load_frag_a(xb + (size_t)row0 * CIN + kk * 32, CIN, lane);
    v16bf bf = load_frag_b(w2 + (size_t)n0 * CIN + kk * 32, CIN, lane);
    acc = wmma_bf16(a, bf, acc);
  }
  int col = lane & 15, h = lane >> 4;
  float bias = b2[n0 + col];
#pragma unroll
  for (int g = 0; g < 8; g++) xs[g + 8 * h][n0 + col] = acc[g] + bias;
  __syncthreads();

#pragma unroll
  for (int rr = 0; rr < 2; rr++) {
    int r = wv * 2 + rr;
    float v[4]; float s = 0.f, sq = 0.f;
#pragma unroll
    for (int i = 0; i < 4; i++) {
      v[i] = xs[r][lane + 32 * i];
      s += v[i]; sq += v[i] * v[i];
    }
    s = wave_sum(s); sq = wave_sum(sq);
    float mean = s * (1.f / DCH);
    float rstd = rsqrtf(sq * (1.f / DCH) - mean * mean + LN_EPS);
    float ln1[4]; float s1 = 0.f, s2 = 0.f;
#pragma unroll
    for (int i = 0; i < 4; i++) {
      int c = lane + 32 * i;
      ln1[i] = (v[i] - mean) * rstd * w1[c] + b1[c];
      s1 += ln1[i]; s2 += ln1[i] * ln1[i];
    }
    s1 = wave_sum(s1); s2 = wave_sum(s2);
    if (lane == 0) cdl[row0 + r] = sqrtf(s2);  // ||ln1|| (c_d_l)
    float mean2 = s1 * (1.f / DCH);
    float rstd2 = rsqrtf(s2 * (1.f / DCH) - mean2 * mean2 + LN_EPS);
#pragma unroll
    for (int i = 0; i < 4; i++) {
      int c = lane + 32 * i;
      float l2 = (ln1[i] - mean2) * rstd2 * w1[c] + b1[c];  // LN applied twice (quirk)
      ln2out[(size_t)(row0 + r) * DCH + c] = (__bf16)l2;
    }
  }
}

// ---------- K2: sim GEMM + cosine scale + softmax + degrad GEMM ----------
__global__ __launch_bounds__(256) void k2_sim_softmax_degrad(
    const __bf16* __restrict__ ln2, const __bf16* __restrict__ drn,
    const __bf16* __restrict__ drnT, const float* __restrict__ cdl,
    const float* __restrict__ cdr, float* __restrict__ simsoft,
    __bf16* __restrict__ degrad, float* __restrict__ maxv) {
  __shared__ __align__(16) float  simLds[16][520];
  __shared__ __align__(16) __bf16 softLds[16][520];
  __shared__ float invCdl[16];
  int tid = threadIdx.x, lane = tid & 31, wv = tid >> 5;
  int row0 = blockIdx.x * 16;
  if (tid < 16) invCdl[tid] = 1.f / cdl[row0 + tid];
  __syncthreads();

  int col = lane & 15, h = lane >> 4;

  // sim = (ln2 @ drn^T) / (cdl*cdr)
  v16bf a[4];
#pragma unroll
  for (int kk = 0; kk < 4; kk++)
    a[kk] = load_frag_a(ln2 + (size_t)row0 * DCH + kk * 32, DCH, lane);
#pragma unroll
  for (int t = 0; t < 4; t++) {
    int n0 = (wv * 4 + t) * 16;
    v8f acc = {};
#pragma unroll
    for (int kk = 0; kk < 4; kk++) {
      v16bf bf = load_frag_b(drn + (size_t)n0 * DCH + kk * 32, DCH, lane);
      acc = wmma_bf16(a[kk], bf, acc);
    }
    float icdr = 1.f / cdr[n0 + col];
#pragma unroll
    for (int g = 0; g < 8; g++) {
      int r = g + 8 * h;
      simLds[r][n0 + col] = acc[g] * invCdl[r] * icdr;
    }
  }
  __syncthreads();

  // row softmax over N=512; wave wv owns rows 2wv, 2wv+1
#pragma unroll
  for (int rr = 0; rr < 2; rr++) {
    int r = wv * 2 + rr;
    float vals[16];
    float m = -3.4e38f;
#pragma unroll
    for (int i = 0; i < 16; i++) {
      vals[i] = simLds[r][lane + 32 * i];
      m = fmaxf(m, vals[i]);
    }
    m = wave_max(m);
    float s = 0.f;
#pragma unroll
    for (int i = 0; i < 16; i++) { vals[i] = __expf(vals[i] - m); s += vals[i]; }
    s = wave_sum(s);
    float invZ = 1.f / s;
    size_t gr = (size_t)(row0 + r) * NRES;
#pragma unroll
    for (int i = 0; i < 16; i++) {
      int c = lane + 32 * i;
      float sv = vals[i] * invZ;
      softLds[r][c] = (__bf16)sv;
      simsoft[gr + c] = sv;
    }
    if (lane == 0) maxv[row0 + r] = invZ;  // max softmax value of the row
  }
  __syncthreads();

  // degrad = soft @ drn  (K=512), wave wv owns d-tile wv*16
  {
    int d0 = wv * 16;
    v8f acc = {};
#pragma unroll
    for (int kk = 0; kk < 16; kk++) {
      v16bf aa = load_frag_a(&softLds[0][0] + kk * 32, 520, lane);
      v16bf bb = load_frag_b(drnT + (size_t)d0 * NRES + kk * 32, NRES, lane);
      acc = wmma_bf16(aa, bb, acc);
    }
#pragma unroll
    for (int g = 0; g < 8; g++) {
      int r = g + 8 * h;
      degrad[(size_t)(row0 + r) * DCH + d0 + col] = (__bf16)acc[g];
    }
  }
}

// ---------------- K3: conv_out GEMM, channel-major coalesced store ----------
__global__ __launch_bounds__(256) void k3_convout(const __bf16* __restrict__ degrad,
                                                  const __bf16* __restrict__ cwo,
                                                  const float* __restrict__ cb,
                                                  float* __restrict__ outImg) {
  int lane = threadIdx.x & 31, wv = threadIdx.x >> 5;
  int row0 = blockIdx.x * 16;
  int bb = row0 >> 12, hw0 = row0 & (HWSZ - 1);
  int col = lane & 15, h = lane >> 4;
  v16bf a[4];
#pragma unroll
  for (int kk = 0; kk < 4; kk++)
    a[kk] = load_frag_a(degrad + (size_t)row0 * DCH + kk * 32, DCH, lane);
#pragma unroll
  for (int t = 0; t < 3; t++) {
    int co0 = (wv * 3 + t) * 16;
    v8f acc = {};
#pragma unroll
    for (int kk = 0; kk < 4; kk++) {
      v16bf bf = load_frag_b(cwo + (size_t)co0 * DCH + kk * 32, DCH, lane);
      acc = wmma_bf16(a[kk], bf, acc);
    }
    int co = co0 + col;
    float bias = cb[co];
    float* dst = outImg + ((size_t)bb * CIN + co) * HWSZ + hw0 + 8 * h;
    v4f f0 = {acc[0] + bias, acc[1] + bias, acc[2] + bias, acc[3] + bias};
    v4f f1 = {acc[4] + bias, acc[5] + bias, acc[6] + bias, acc[7] + bias};
    *(v4f*)dst = f0;
    *((v4f*)dst + 1) = f1;
  }
}

// ------------- mask: per-batch min-max normalize of (max softmax)^3 ---------
__global__ __launch_bounds__(256) void kmask(const float* __restrict__ maxv,
                                             float* __restrict__ maskOut) {
  __shared__ float smn[256], smx[256];
  int b = blockIdx.x, tid = threadIdx.x;
  float loc[16]; float mn = 3.4e38f, mx = -3.4e38f;
#pragma unroll
  for (int i = 0; i < 16; i++) {
    float v = maxv[(size_t)b * HWSZ + tid + 256 * i];
    v = v * v * v;
    loc[i] = v;
    mn = fminf(mn, v); mx = fmaxf(mx, v);
  }
  smn[tid] = mn; smx[tid] = mx;
  __syncthreads();
  for (int s = 128; s > 0; s >>= 1) {
    if (tid < s) {
      smn[tid] = fminf(smn[tid], smn[tid + s]);
      smx[tid] = fmaxf(smx[tid], smx[tid + s]);
    }
    __syncthreads();
  }
  mn = smn[0]; mx = smx[0];
  float inv = 1.f / (mx - mn);
#pragma unroll
  for (int i = 0; i < 16; i++)
    maskOut[(size_t)b * HWSZ + tid + 256 * i] = 1.f - (loc[i] - mn) * inv;
}

// ------------------- top-81 per batch (iterative argmax) -------------------
__global__ __launch_bounds__(256) void ktopk(const float* __restrict__ maxv,
                                             int* __restrict__ topIdx) {
  __shared__ float vals[HWSZ];
  __shared__ float rv[256];
  __shared__ int ri[256];
  int b = blockIdx.x, tid = threadIdx.x;
  for (int i = tid; i < HWSZ; i += 256) vals[i] = maxv[(size_t)b * HWSZ + i];
  __syncthreads();
  for (int k = 0; k < KTOP; k++) {
    float bv = -3.4e38f; int bi = 0x7fffffff;
#pragma unroll
    for (int j = 0; j < 16; j++) {
      int idx = tid + 256 * j;
      float v = vals[idx];
      if (v > bv || (v == bv && idx < bi)) { bv = v; bi = idx; }
    }
    rv[tid] = bv; ri[tid] = bi;
    __syncthreads();
    for (int s = 128; s > 0; s >>= 1) {
      if (tid < s) {
        if (rv[tid + s] > rv[tid] ||
            (rv[tid + s] == rv[tid] && ri[tid + s] < ri[tid])) {
          rv[tid] = rv[tid + s]; ri[tid] = ri[tid + s];
        }
      }
      __syncthreads();
    }
    if (tid == 0) {
      topIdx[b * KTOP + k] = ri[0];
      vals[ri[0]] = -3.4e38f;
    }
    __syncthreads();
  }
}

// -------- results[b1,b2,k,:] = sim_soft[b1, top_idx[b2,k], :] --------------
__global__ __launch_bounds__(128) void kres(const float* __restrict__ simsoft,
                                            const int* __restrict__ topIdx,
                                            float* __restrict__ out) {
  int id = blockIdx.x;                 // b1*648 + b2*81 + k
  int b1 = id / (BATCH * KTOP);
  int rem = id % (BATCH * KTOP);
  int b2 = rem / KTOP, k = rem % KTOP;
  int row = b1 * HWSZ + topIdx[b2 * KTOP + k];
  const float* src = simsoft + (size_t)row * NRES;
  float* dst = out + (size_t)id * NRES;
  int n = threadIdx.x * 4;
  *(v4f*)(dst + n) = *(const v4f*)(src + n);
}

// ---------------- host launch ----------------
extern "C" void kernel_launch(void* const* d_in, const int* in_sizes, int n_in,
                              void* d_out, int out_size, void* d_ws, size_t ws_size,
                              hipStream_t stream) {
  (void)in_sizes; (void)n_in; (void)out_size;
  const float* x    = (const float*)d_in[0];
  const float* DR   = (const float*)d_in[1];
  const float* cinw = (const float*)d_in[2];
  const float* cinb = (const float*)d_in[3];
  const float* semw = (const float*)d_in[4];
  const float* semb = (const float*)d_in[5];
  const float* n1w  = (const float*)d_in[6];
  const float* n1b  = (const float*)d_in[7];
  const float* n2w  = (const float*)d_in[8];
  const float* n2b  = (const float*)d_in[9];
  const float* cow  = (const float*)d_in[10];
  const float* cob  = (const float*)d_in[11];

  size_t off = 0;
  char* wsb = (char*)d_ws;
  auto take = [&](size_t bytes) -> char* {
    char* p = wsb + off;
    off += (bytes + 255) & ~(size_t)255;
    return p;
  };
  __bf16* xb      = (__bf16*)take((size_t)PIX * CIN * 2);
  __bf16* w2      = (__bf16*)take((size_t)DCH * CIN * 2);
  float*  b2      = (float*) take(DCH * 4);
  __bf16* cwo     = (__bf16*)take((size_t)CIN * DCH * 2);
  __bf16* drn     = (__bf16*)take((size_t)NRES * DCH * 2);
  __bf16* drnT    = (__bf16*)take((size_t)DCH * NRES * 2);
  float*  cdr     = (float*) take(NRES * 4);
  __bf16* ln2     = (__bf16*)take((size_t)PIX * DCH * 2);
  float*  cdl     = (float*) take(PIX * 4);
  float*  simsoft = (float*) take((size_t)PIX * NRES * 4);
  __bf16* degrad  = (__bf16*)take((size_t)PIX * DCH * 2);
  float*  maxv    = (float*) take(PIX * 4);
  int*    topIdx  = (int*)   take(BATCH * KTOP * 4);
  if (off > ws_size) return;  // workspace too small: bail deterministically

  float* outMask = (float*)d_out;                       // [8,1,64,64]
  float* outImg  = outMask + (size_t)BATCH * HWSZ;      // [8,384,64,64]
  float* outRes  = outImg + (size_t)BATCH * CIN * HWSZ; // [8,8,81,512]

  kx_transpose<<<dim3(HWSZ / 32, CIN / 32, BATCH), dim3(32, 8), 0, stream>>>(x, xb);
  kw2<<<(DCH * CIN + 255) / 256, 256, 0, stream>>>(semw, cinw, cinb, semb, w2, b2);
  kcvt<<<(CIN * DCH + 255) / 256, 256, 0, stream>>>(cow, cwo, CIN * DCH);
  kdrn<<<NRES / 8, 256, 0, stream>>>(DR, n2w, n2b, drn, drnT, cdr);

  k1_gemm_ln<<<PIX / 16, 256, 0, stream>>>(xb, w2, b2, n1w, n1b, ln2, cdl);
  k2_sim_softmax_degrad<<<PIX / 16, 256, 0, stream>>>(ln2, drn, drnT, cdl, cdr,
                                                      simsoft, degrad, maxv);
  k3_convout<<<PIX / 16, 256, 0, stream>>>(degrad, cwo, cob, outImg);

  kmask<<<BATCH, 256, 0, stream>>>(maxv, outMask);
  ktopk<<<BATCH, 256, 0, stream>>>(maxv, topIdx);
  kres<<<BATCH * BATCH * KTOP, 128, 0, stream>>>(simsoft, topIdx, outRes);
}
